// RelationNetworks_87754771792412
// MI455X (gfx1250) — matrix operand order repaired and checked by reading, not compiled, not run on hardware
//
#include <hip/hip_runtime.h>
#include <math.h>

typedef __attribute__((ext_vector_type(16))) _Float16 v16h;
typedef __attribute__((ext_vector_type(8)))  float    v8f;

#define BSZ 8
#define MXN 24
#define TT 30
#define EMB 32
#define LH 128
#define CLS 29
#define BN_IMG (BSZ*MXN)   // 192

__device__ __forceinline__ float relu_f(float x){ return x > 0.f ? x : 0.f; }

// ---------------- generic conv + relu (NCHW / OIHW) ----------------
__global__ void k_conv_relu(const float* __restrict__ x, const float* __restrict__ w,
                            const float* __restrict__ bias, float* __restrict__ y,
                            int NBt, int Cin, int Hin, int Win, int Cout, int K,
                            int stride, int pad, int Hout, int Wout)
{
  int idx = blockIdx.x * blockDim.x + threadIdx.x;
  int total = NBt * Cout * Hout * Wout;
  if (idx >= total) return;
  int wo = idx % Wout; int t = idx / Wout;
  int ho = t % Hout;   t /= Hout;
  int co = t % Cout;   int n = t / Cout;
  float acc = bias[co];
  for (int ci = 0; ci < Cin; ++ci){
    const float* xp = x + (size_t)((n * Cin + ci) * Hin) * Win;
    const float* wp = w + (size_t)((co * Cin + ci) * K) * K;
    for (int kh = 0; kh < K; ++kh){
      int hh = ho * stride - pad + kh;
      if (hh < 0 || hh >= Hin) continue;
      for (int kw = 0; kw < K; ++kw){
        int ww = wo * stride - pad + kw;
        if (ww < 0 || ww >= Win) continue;
        acc += xp[hh * Win + ww] * wp[kh * K + kw];
      }
    }
  }
  y[idx] = relu_f(acc);
}

// ---------------- batch-norm stats: per-channel mean / rstd ----------------
__global__ void k_bn_stats(const float* __restrict__ y, float* __restrict__ meanv,
                           float* __restrict__ rstdv, int NBt, int C, int HW)
{
  int c = blockIdx.x;
  __shared__ float s0[256], s1[256];
  float sum = 0.f, sq = 0.f;
  int cnt = NBt * HW;
  for (int i = threadIdx.x; i < cnt; i += blockDim.x){
    int n = i / HW, s = i % HW;
    float v = y[(size_t)(n * C + c) * HW + s];
    sum += v; sq += v * v;
  }
  s0[threadIdx.x] = sum; s1[threadIdx.x] = sq;
  __syncthreads();
  for (int off = 128; off > 0; off >>= 1){
    if ((int)threadIdx.x < off){
      s0[threadIdx.x] += s0[threadIdx.x + off];
      s1[threadIdx.x] += s1[threadIdx.x + off];
    }
    __syncthreads();
  }
  if (threadIdx.x == 0){
    float m = s0[0] / (float)cnt;
    float v = s1[0] / (float)cnt - m * m;
    meanv[c] = m;
    rstdv[c] = rsqrtf(v + 1e-5f);
  }
}

__global__ void k_bn_apply(float* __restrict__ y, const float* __restrict__ meanv,
                           const float* __restrict__ rstdv, int total, int C, int HW)
{
  int idx = blockIdx.x * blockDim.x + threadIdx.x;
  if (idx >= total) return;
  int c = (idx / HW) % C;
  y[idx] = (y[idx] - meanv[c]) * rstdv[c];
}

// stage-3 (1x1) outputs normalized straight into x_all[:, off:off+16]
__global__ void k_bn_apply_xall(const float* __restrict__ tmp, const float* __restrict__ meanv,
                                const float* __restrict__ rstdv, float* __restrict__ x_all, int off)
{
  int idx = blockIdx.x * blockDim.x + threadIdx.x; // BN_IMG*16
  if (idx >= BN_IMG * 16) return;
  int c = idx % 16, n = idx / 16;
  x_all[n * 32 + off + c] = (tmp[idx] - meanv[c]) * rstdv[c];
}

// ---------------- 2x2 max-pool ----------------
__global__ void k_pool(const float* __restrict__ x, float* __restrict__ y,
                       int NBt, int C, int Hin, int Win)
{
  int Hout = Hin / 2, Wout = Win / 2;
  int idx = blockIdx.x * blockDim.x + threadIdx.x;
  int total = NBt * C * Hout * Wout;
  if (idx >= total) return;
  int wo = idx % Wout; int t = idx / Wout;
  int ho = t % Hout;   t /= Hout;
  int cc = t % C;      int n = t / C;
  const float* xp = x + (size_t)((n * C + cc) * Hin + ho * 2) * Win + wo * 2;
  y[idx] = fmaxf(fmaxf(xp[0], xp[1]), fmaxf(xp[Win], xp[Win + 1]));
}

// ---------------- LSTM over question (single block, h/c/gates in LDS) ----------------
__global__ void __launch_bounds__(512)
k_lstm(const float* __restrict__ embed_w, const int* __restrict__ question,
       const int* __restrict__ qlen,
       const float* __restrict__ wih, const float* __restrict__ whh,
       const float* __restrict__ bih, const float* __restrict__ bhh,
       float* __restrict__ qst)
{
  __shared__ float h[BSZ][LH], c[BSZ][LH], g[BSZ][4 * LH], xt[BSZ][EMB];
  int tid = threadIdx.x;
  for (int i = tid; i < BSZ * LH; i += 512){ h[i / LH][i % LH] = 0.f; c[i / LH][i % LH] = 0.f; }
  __syncthreads();
  for (int t = 0; t < TT; ++t){
    for (int i = tid; i < BSZ * EMB; i += 512){
      int b = i / EMB, k = i % EMB;
      xt[b][k] = embed_w[question[b * TT + t] * EMB + k];
    }
    __syncthreads();
    // gate j for all batches
    {
      int j = tid;   // 0..511
      float bsum = bih[j] + bhh[j];
      const float* wi = wih + j * EMB;
      const float* wh = whh + j * LH;
      for (int b = 0; b < BSZ; ++b){
        float acc = bsum;
        for (int k = 0; k < EMB; ++k) acc += xt[b][k] * wi[k];
        for (int k = 0; k < LH;  ++k) acc += h[b][k] * wh[k];
        g[b][j] = acc;
      }
    }
    __syncthreads();
    for (int i = tid; i < BSZ * LH; i += 512){
      int b = i / LH, jj = i % LH;
      float gi = g[b][jj], gf = g[b][LH + jj], gg = g[b][2 * LH + jj], go = g[b][3 * LH + jj];
      float i_ = 1.f / (1.f + expf(-gi));
      float f_ = 1.f / (1.f + expf(-gf));
      float g_ = tanhf(gg);
      float o_ = 1.f / (1.f + expf(-go));
      float cn = f_ * c[b][jj] + i_ * g_;
      float hn = o_ * tanhf(cn);
      if (t < qlen[b]) { h[b][jj] = hn; c[b][jj] = cn; }
    }
    __syncthreads();
  }
  for (int i = tid; i < BSZ * LH; i += 512) qst[i] = h[i / LH][i % LH];
}

// ---------------- small GEMV: out[r,d] = sum_k in[r,k]*W[d, off+k] (+bias[d]) ----------------
__global__ void k_gemv(const float* __restrict__ in, const float* __restrict__ W,
                       const float* __restrict__ bias, float* __restrict__ out,
                       int rows, int inStride, int Kd, int wStride, int wOff, int D)
{
  int idx = blockIdx.x * blockDim.x + threadIdx.x;
  if (idx >= rows * D) return;
  int d = idx % D, r = idx / D;
  const float* ip = in + (size_t)r * inStride;
  const float* wp = W + (size_t)d * wStride + wOff;
  float acc = bias ? bias[d] : 0.f;
  for (int k = 0; k < Kd; ++k) acc += ip[k] * wp[k];
  out[idx] = acc;
}

__global__ void k_tohalf(const float* __restrict__ in, _Float16* __restrict__ out, int n)
{
  int i = blockIdx.x * blockDim.x + threadIdx.x;
  if (i < n) out[i] = (_Float16)in[i];
}

__global__ void k_zero(float* __restrict__ p, int n)
{
  int i = blockIdx.x * blockDim.x + threadIdx.x;
  if (i < n) p[i] = 0.f;
}

// ---------------- fused relation GEMM (WMMA f16 -> f32) ----------------
// One block per (b, outer). Rows = last index r (24 used, padded to 32).
// u[r,k] = relu(base[b,k] + prefA[k] (+ prefB[k]) + rowterm[b,r,k]); out = relu(u @ Wh^T + b4);
// vout[b,d] += sum_r rowmask[r] * out[r,d].
__global__ void __launch_bounds__(256)
k_relgemm(const float* __restrict__ base,     // [B,256] (hq + b1)
          const float* __restrict__ prefA,    // [B,N,256]
          const float* __restrict__ prefB,    // [B,N,256] or unused
          const float* __restrict__ rowterm,  // [B,N,256]
          const _Float16* __restrict__ Wh,    // [256,256] row-major (d-major): B[k][d]=Wh[d*256+k]
          const float* __restrict__ b4,
          const int* __restrict__ num_layers,
          float* __restrict__ vout,           // [B,256] accumulated atomically
          int nOuter, int twopref)
{
  int blk = blockIdx.x;
  int b = blk / nOuter;
  int outer = blk % nOuter;
  int nl = num_layers[b]; if (nl > MXN) nl = MXN;

  const float* pA;
  const float* pB = nullptr;
  if (twopref){
    int p = outer / MXN, q = outer % MXN;
    if (p >= nl || q >= nl) return;            // whole (p,q) slab masked out
    pA = prefA + (size_t)(b * MXN + p) * 256;
    pB = prefB + (size_t)(b * MXN + q) * 256;
  } else {
    if (outer >= nl) return;
    pA = prefA + (size_t)(b * MXN + outer) * 256;
  }

  __shared__ _Float16 U[32][264];   // 32 rows (24 real + 8 zero pad), 256 cols (+pad)
  __shared__ float rm[32];
  const float* bs = base + b * 256;

  for (int idx = threadIdx.x; idx < 32 * 256; idx += 256){
    int r = idx >> 8, k = idx & 255;
    float v = 0.f;
    if (r < MXN){
      v = bs[k] + pA[k] + rowterm[(size_t)(b * MXN + r) * 256 + k];
      if (pB) v += pB[k];
      v = relu_f(v);                 // inner relu
    }
    U[r][k] = (_Float16)v;
  }
  if (threadIdx.x < 32) rm[threadIdx.x] = ((int)threadIdx.x < nl) ? 1.f : 0.f;
  __syncthreads();

  const int lane = threadIdx.x & 31;
  const int wave = threadIdx.x >> 5;   // 8 waves: wave handles N-tiles 2w, 2w+1
  const int grp  = lane >> 4;
  const int lm   = lane & 15;

  v8f acc[2][2] = {};                  // [mtile][ntile-in-wave]

  #pragma unroll 1
  for (int kk = 0; kk < 8; ++kk){
    // A fragments: contiguous K to match B (lanes 0-15: K 0-15; lanes 16-31: K 16-31)
    v16h a[2];
    #pragma unroll
    for (int mt = 0; mt < 2; ++mt){
      int m = mt * 16 + lm;
      #pragma unroll
      for (int h = 0; h < 16; ++h){
        int vg = h >> 1, w = h & 1;
        int k = kk * 32 + (grp << 4) + (vg << 1) + w;
        a[mt][h] = U[m][k];
      }
    }
    #pragma unroll
    for (int nti = 0; nti < 2; ++nti){
      int d = (wave * 2 + nti) * 16 + lm;
      // B fragment: lanes 0-15 hold K=kk*32..+15, lanes 16-31 K=kk*32+16..+31; contiguous 32B
      v16h bfrag = *(const v16h*)(Wh + (size_t)d * 256 + kk * 32 + (grp << 4));
      acc[0][nti] = __builtin_amdgcn_wmma_f32_16x16x32_f16(
          false, a[0], false, bfrag, (short)0, acc[0][nti], false, false);
      acc[1][nti] = __builtin_amdgcn_wmma_f32_16x16x32_f16(
          false, a[1], false, bfrag, (short)0, acc[1][nti], false, false);
    }
  }

  // bias + outer relu + row-mask + column reduction, atomically into vout[b,:]
  #pragma unroll
  for (int nti = 0; nti < 2; ++nti){
    int d = (wave * 2 + nti) * 16 + lm;
    float bias = b4[d];
    float colsum = 0.f;
    #pragma unroll
    for (int mt = 0; mt < 2; ++mt){
      #pragma unroll
      for (int rr = 0; rr < 8; ++rr){
        int m = mt * 16 + grp * 8 + rr;       // C/D layout: VGPR rr -> M = rr + 8*grp
        float x = acc[mt][nti][rr] + bias;
        colsum += rm[m] * relu_f(x);
      }
    }
    atomicAdd(&vout[b * 256 + d], colsum);
  }
}

// ---------------- final MLP + log_softmax ----------------
__global__ void __launch_bounds__(256)
k_final(const float* __restrict__ v2, const float* __restrict__ v3,
        const float* __restrict__ w1, const float* __restrict__ b1,
        const float* __restrict__ w2, const float* __restrict__ b2,
        const float* __restrict__ w3, const float* __restrict__ b3,
        float* __restrict__ out)
{
  __shared__ float xg[512], h1[256], h2[256], lg[CLS], red[2];
  int tid = threadIdx.x;
  for (int b = 0; b < BSZ; ++b){
    for (int i = tid; i < 512; i += 256)
      xg[i] = (i < 256) ? v2[b * 256 + i] : v3[b * 256 + (i - 256)];
    __syncthreads();
    float a = b1[tid];
    for (int k = 0; k < 512; ++k) a += xg[k] * w1[tid * 512 + k];
    h1[tid] = relu_f(a);
    __syncthreads();
    a = b2[tid];
    for (int k = 0; k < 256; ++k) a += h1[k] * w2[tid * 256 + k];
    h2[tid] = relu_f(a);
    __syncthreads();
    if (tid < CLS){
      a = b3[tid];
      for (int k = 0; k < 256; ++k) a += h2[k] * w3[tid * 256 + k];
      lg[tid] = a;
    }
    __syncthreads();
    if (tid == 0){
      float mx = lg[0];
      for (int i = 1; i < CLS; ++i) mx = fmaxf(mx, lg[i]);
      float s = 0.f;
      for (int i = 0; i < CLS; ++i) s += expf(lg[i] - mx);
      red[0] = mx; red[1] = logf(s);
    }
    __syncthreads();
    if (tid < CLS) out[b * CLS + tid] = lg[tid] - red[0] - red[1];
    __syncthreads();
  }
}

// =======================================================================
extern "C" void kernel_launch(void* const* d_in, const int* in_sizes, int n_in,
                              void* d_out, int out_size, void* d_ws, size_t ws_size,
                              hipStream_t stream)
{
  (void)in_sizes; (void)n_in; (void)out_size; (void)ws_size;

  const float* apps        = (const float*)d_in[0];
  const float* masks       = (const float*)d_in[1];
  const int*   num_layers  = (const int*)  d_in[2];
  const int*   question    = (const int*)  d_in[3];
  const int*   qlen        = (const int*)  d_in[4];
  const float* ma_w1 = (const float*)d_in[5];  const float* ma_b1 = (const float*)d_in[6];
  const float* ma_w2 = (const float*)d_in[7];  const float* ma_b2 = (const float*)d_in[8];
  const float* ma_w3 = (const float*)d_in[9];  const float* ma_b3 = (const float*)d_in[10];
  const float* sg_w1 = (const float*)d_in[11]; const float* sg_b1 = (const float*)d_in[12];
  const float* sg_w2 = (const float*)d_in[13]; const float* sg_b2 = (const float*)d_in[14];
  const float* sg_w3 = (const float*)d_in[15]; const float* sg_b3 = (const float*)d_in[16];
  const float* g2_w1 = (const float*)d_in[17]; const float* g2_b1 = (const float*)d_in[18];
  const float* g2_w4 = (const float*)d_in[19]; const float* g2_b4 = (const float*)d_in[20];
  const float* g3_w1 = (const float*)d_in[21]; const float* g3_b1 = (const float*)d_in[22];
  const float* g3_w4 = (const float*)d_in[23]; const float* g3_b4 = (const float*)d_in[24];
  const float* f_w1  = (const float*)d_in[25]; const float* f_b1  = (const float*)d_in[26];
  const float* f_w2  = (const float*)d_in[27]; const float* f_b2  = (const float*)d_in[28];
  const float* f_w3  = (const float*)d_in[29]; const float* f_b3  = (const float*)d_in[30];
  const float* emb_w = (const float*)d_in[31];
  const float* l_wih = (const float*)d_in[32]; const float* l_whh = (const float*)d_in[33];
  const float* l_bih = (const float*)d_in[34]; const float* l_bhh = (const float*)d_in[35];

  float* W = (float*)d_ws;
  size_t off = 0;
  auto alloc = [&](size_t n) { size_t o = off; off += (n + 63) & ~size_t(63); return o; };

  const size_t o_m1  = alloc(192u*24*16*16);
  const size_t o_m2  = alloc(192u*32*8*8);
  const size_t o_m3t = alloc(192u*16);
  const size_t o_a1  = alloc(192u*16*32*32);
  const size_t o_a1p = alloc(192u*16*16*16);
  const size_t o_a2  = alloc(192u*16*16*16);
  const size_t o_a2p = alloc(192u*16*8*8);
  const size_t o_a3t = alloc(192u*16);
  const size_t o_xall= alloc(192u*32);
  const size_t o_qst = alloc(BSZ*LH);
  const size_t o_hi  = alloc(192u*256);
  const size_t o_hj  = alloc(192u*256);
  const size_t o_hq2 = alloc(BSZ*256);
  const size_t o_hi3 = alloc(192u*256);
  const size_t o_hj3 = alloc(192u*256);
  const size_t o_hk3 = alloc(192u*256);
  const size_t o_hq3 = alloc(BSZ*256);
  const size_t o_v2  = alloc(BSZ*256);
  const size_t o_v3  = alloc(BSZ*256);
  const size_t o_mean= alloc(64);
  const size_t o_rstd= alloc(64);
  const size_t o_w2h = alloc(256u*256/2);   // f16 storage in float units
  const size_t o_w3h = alloc(256u*256/2);

  _Float16* w2h = (_Float16*)(W + o_w2h);
  _Float16* w3h = (_Float16*)(W + o_w3h);
  float* meanv = W + o_mean; float* rstdv = W + o_rstd;

  auto NBLK = [](int n){ return (n + 255) / 256; };

  // ---- mask conv stack: 1->24 (5x5 s2 p2, 16x16), 24->32 (5x5 s2 p2, 8x8), 32->16 (8x8, 1x1)
  {
    int tot = 192*24*16*16;
    k_conv_relu<<<NBLK(tot),256,0,stream>>>(masks, ma_w1, ma_b1, W+o_m1, 192,1,32,32,24,5,2,2,16,16);
    k_bn_stats<<<24,256,0,stream>>>(W+o_m1, meanv, rstdv, 192,24,256);
    k_bn_apply<<<NBLK(tot),256,0,stream>>>(W+o_m1, meanv, rstdv, tot, 24, 256);
  }
  {
    int tot = 192*32*8*8;
    k_conv_relu<<<NBLK(tot),256,0,stream>>>(W+o_m1, ma_w2, ma_b2, W+o_m2, 192,24,16,16,32,5,2,2,8,8);
    k_bn_stats<<<32,256,0,stream>>>(W+o_m2, meanv, rstdv, 192,32,64);
    k_bn_apply<<<NBLK(tot),256,0,stream>>>(W+o_m2, meanv, rstdv, tot, 32, 64);
  }
  {
    int tot = 192*16;
    k_conv_relu<<<NBLK(tot),256,0,stream>>>(W+o_m2, ma_w3, ma_b3, W+o_m3t, 192,32,8,8,16,8,1,0,1,1);
    k_bn_stats<<<16,256,0,stream>>>(W+o_m3t, meanv, rstdv, 192,16,1);
    k_bn_apply_xall<<<NBLK(tot),256,0,stream>>>(W+o_m3t, meanv, rstdv, W+o_xall, 16);
  }
  // ---- app conv stack: 3->16 (3x3 p1) bn pool, 16->16 (3x3 p1) bn pool, 16->16 (8x8, 1x1)
  {
    int tot = 192*16*32*32;
    k_conv_relu<<<NBLK(tot),256,0,stream>>>(apps, sg_w1, sg_b1, W+o_a1, 192,3,32,32,16,3,1,1,32,32);
    k_bn_stats<<<16,256,0,stream>>>(W+o_a1, meanv, rstdv, 192,16,1024);
    k_bn_apply<<<NBLK(tot),256,0,stream>>>(W+o_a1, meanv, rstdv, tot, 16, 1024);
    k_pool<<<NBLK(192*16*16*16),256,0,stream>>>(W+o_a1, W+o_a1p, 192,16,32,32);
  }
  {
    int tot = 192*16*16*16;
    k_conv_relu<<<NBLK(tot),256,0,stream>>>(W+o_a1p, sg_w2, sg_b2, W+o_a2, 192,16,16,16,16,3,1,1,16,16);
    k_bn_stats<<<16,256,0,stream>>>(W+o_a2, meanv, rstdv, 192,16,256);
    k_bn_apply<<<NBLK(tot),256,0,stream>>>(W+o_a2, meanv, rstdv, tot, 16, 256);
    k_pool<<<NBLK(192*16*8*8),256,0,stream>>>(W+o_a2, W+o_a2p, 192,16,16,16);
  }
  {
    int tot = 192*16;
    k_conv_relu<<<NBLK(tot),256,0,stream>>>(W+o_a2p, sg_w3, sg_b3, W+o_a3t, 192,16,8,8,16,8,1,0,1,1);
    k_bn_stats<<<16,256,0,stream>>>(W+o_a3t, meanv, rstdv, 192,16,1);
    k_bn_apply_xall<<<NBLK(tot),256,0,stream>>>(W+o_a3t, meanv, rstdv, W+o_xall, 0);
  }

  // ---- LSTM -> qst [8,128]
  k_lstm<<<1,512,0,stream>>>(emb_w, question, qlen, l_wih, l_whh, l_bih, l_bhh, W+o_qst);

  // ---- projections (b1 biases folded into hq terms)
  k_gemv<<<NBLK(192*256),256,0,stream>>>(W+o_xall, g2_w1, nullptr, W+o_hi,  192,32,32,192,0,  256);
  k_gemv<<<NBLK(192*256),256,0,stream>>>(W+o_xall, g2_w1, nullptr, W+o_hj,  192,32,32,192,32, 256);
  k_gemv<<<NBLK(8*256),  256,0,stream>>>(W+o_qst,  g2_w1, g2_b1,   W+o_hq2, 8,128,128,192,64, 256);
  k_gemv<<<NBLK(192*256),256,0,stream>>>(W+o_xall, g3_w1, nullptr, W+o_hi3, 192,32,32,224,0,  256);
  k_gemv<<<NBLK(192*256),256,0,stream>>>(W+o_xall, g3_w1, nullptr, W+o_hj3, 192,32,32,224,32, 256);
  k_gemv<<<NBLK(192*256),256,0,stream>>>(W+o_xall, g3_w1, nullptr, W+o_hk3, 192,32,32,224,64, 256);
  k_gemv<<<NBLK(8*256),  256,0,stream>>>(W+o_qst,  g3_w1, g3_b1,   W+o_hq3, 8,128,128,224,96, 256);

  // ---- f16 weight copies, zero accumulators
  k_tohalf<<<NBLK(65536),256,0,stream>>>(g2_w4, w2h, 65536);
  k_tohalf<<<NBLK(65536),256,0,stream>>>(g3_w4, w3h, 65536);
  k_zero<<<NBLK(BSZ*256),256,0,stream>>>(W+o_v2, BSZ*256);
  k_zero<<<NBLK(BSZ*256),256,0,stream>>>(W+o_v3, BSZ*256);

  // ---- fused relation GEMMs (WMMA)
  // g2: pre2[b,a,c] = hj[b,a] + hi[b,c] + hq2[b]
  k_relgemm<<<BSZ*MXN,256,0,stream>>>(W+o_hq2, W+o_hj, nullptr, W+o_hi,
                                      w2h, g2_b4, num_layers, W+o_v2, MXN, 0);
  // g3: pre3[b,p,q,r] = hk3[b,p] + hj3[b,q] + hi3[b,r] + hq3[b]
  k_relgemm<<<BSZ*MXN*MXN,256,0,stream>>>(W+o_hq3, W+o_hk3, W+o_hj3, W+o_hi3,
                                          w3h, g3_b4, num_layers, W+o_v3, MXN*MXN, 1);

  // ---- final MLP + log_softmax -> d_out [8,29] f32
  k_final<<<1,256,0,stream>>>(W+o_v2, W+o_v3, f_w1, f_b1, f_w2, f_b2, f_w3, f_b3, (float*)d_out);
}
